// FlatQuantAttention_41626823033084
// MI455X (gfx1250) — matrix operand order repaired and checked by reading, not compiled, not run on hardware
//
#include <hip/hip_runtime.h>
#include <math.h>
#include <stdint.h>

// ---------------- types ----------------
typedef __bf16 bf16;
typedef __attribute__((ext_vector_type(16))) __bf16 v16bf;
typedef __attribute__((ext_vector_type(8)))  __bf16 v8bf;
typedef __attribute__((ext_vector_type(8)))  float  v8f;

#define WMMA_BF16(a, b, c) \
  __builtin_amdgcn_wmma_f32_16x16x32_bf16(false, (a), false, (b), (short)0, (c), false, false)

// ---------------- constants ----------------
#define S_LEN 2048
#define DM    4096
#define NHEAD 32
#define NKVH  8
#define HD    128

// ---------------- CDNA5 async global->LDS copy (ISA 10.7 / 15.18.3) ----------
// Per-lane: LDS[lds_off] = MEM[gaddr], 16 bytes, tracked by ASYNCcnt.
__device__ __forceinline__ void async_load_b128(uint32_t lds_off, const void* gaddr) {
  asm volatile("global_load_async_to_lds_b128 %0, %1, off"
               :: "v"(lds_off), "v"(gaddr)
               : "memory");
}
__device__ __forceinline__ void wait_asynccnt0() {
  asm volatile("s_wait_asynccnt 0x0" ::: "memory");
}
// Generic shared-pointer -> 32-bit LDS offset (ISA 10.2: LDS offset is the low 32 address bits).
__device__ __forceinline__ uint32_t lds_off32(const void* p) {
  return (uint32_t)(uintptr_t)p;
}

// ---------------- WMMA fragment loaders (wave32 layouts, CDNA5 ISA 7.12.2) ----
// A: 16x32 bf16 (MxK) from a row-major tile (K contiguous): two ds_load_b128.
__device__ __forceinline__ v16bf lds_load_A(const bf16* base, int ld, int row0, int k0) {
  int lane = threadIdx.x & 31;
  int m  = row0 + (lane & 15);
  int kb = (lane & 16) ? 8 : 0;
  const bf16* r = base + (size_t)m * ld + k0 + kb;
  v8bf lo = *(const v8bf*)(r);        // K = kb+0 .. kb+7
  v8bf hi = *(const v8bf*)(r + 16);   // K = 16+kb+0 .. 16+kb+7
  return __builtin_shufflevector(lo, hi, 0, 1, 2, 3, 4, 5, 6, 7,
                                 8, 9, 10, 11, 12, 13, 14, 15);
}

// B: 32x16 bf16 (KxN) from an N-major tile base[n*ldk + k]: two ds_load_b128.
__device__ __forceinline__ v16bf lds_load_Bt(const bf16* base, int ldk, int k0, int n0) {
  int lane = threadIdx.x & 31;
  int n  = lane & 15;
  int kb = (lane & 16) ? 8 : 0;
  const bf16* r = base + (size_t)(n0 + n) * ldk + k0 + kb;
  v8bf lo = *(const v8bf*)(r);
  v8bf hi = *(const v8bf*)(r + 16);
  return __builtin_shufflevector(lo, hi, 0, 1, 2, 3, 4, 5, 6, 7,
                                 8, 9, 10, 11, 12, 13, 14, 15);
}

// ---------------- conversion kernels ----------------
// out[c*rows + r] = in[r*cols + c]  (bf16 transpose of an f32 matrix)
__global__ void k_f2bfT(const float* __restrict__ in, bf16* __restrict__ out, int rows, int cols) {
  int i = blockIdx.x * blockDim.x + threadIdx.x;
  if (i < rows * cols) {
    int r = i / cols, c = i % cols;
    out[c * rows + r] = (bf16)in[i];
  }
}

// ---------------- 128x128 Gauss-Jordan inverse (single block, 128 threads) ----
__global__ __launch_bounds__(128) void k_invert128(const float* __restrict__ T,
                                                   float* __restrict__ inv,
                                                   float* __restrict__ work) {
  int t = threadIdx.x;
  for (int c = 0; c < 128; ++c) {
    work[t * 128 + c] = T[t * 128 + c];
    inv[t * 128 + c]  = (t == c) ? 1.0f : 0.0f;
  }
  __syncthreads();
  for (int p = 0; p < 128; ++p) {
    if (t == p) {
      float ip = 1.0f / work[p * 128 + p];
      for (int c = 0; c < 128; ++c) { work[p * 128 + c] *= ip; inv[p * 128 + c] *= ip; }
    }
    __syncthreads();
    if (t != p) {
      float f = work[t * 128 + p];
      for (int c = 0; c < 128; ++c) {
        work[t * 128 + c] -= f * work[p * 128 + c];
        inv[t * 128 + c]  -= f * inv[p * 128 + c];
      }
    }
    __syncthreads();
  }
}

// ---------------- per-token 64x64 transform: out = L^T * h * R -----------------
// Lt_bf = ln_left^T (row-major, A-side), Rt_bf = ln_right^T (N-major B tile).
__global__ __launch_bounds__(128) void k_token_transform(const float* __restrict__ hidden,
                                                         const bf16* __restrict__ Lt,
                                                         const bf16* __restrict__ Rt,
                                                         bf16* __restrict__ out) {
  __shared__ __align__(16) bf16 sH[64 * 64];
  __shared__ __align__(16) bf16 sRt[64 * 64];
  __shared__ __align__(16) bf16 sL[64 * 64];
  __shared__ __align__(16) bf16 sTt[64 * 64];  // N-major: sTt[n*64 + k] = T[k][n]
  int tok = blockIdx.x;
  const float* hp = hidden + (size_t)tok * DM;
  for (int i = threadIdx.x; i < 4096; i += 128) sH[i] = (bf16)hp[i];
  for (int i = threadIdx.x; i < 512; i += 128) {
    *(v8bf*)&sRt[i * 8] = *(const v8bf*)&Rt[i * 8];
    *(v8bf*)&sL[i * 8]  = *(const v8bf*)&Lt[i * 8];
  }
  __syncthreads();
  int w = threadIdx.x >> 5;
  int row0 = w * 16;
  int lane = threadIdx.x & 31;
  int col = lane & 15;
  int hb = (lane & 16) ? 8 : 0;

  // T = H @ R
  v8f acc[4] = {};
#pragma unroll
  for (int kc = 0; kc < 2; ++kc) {
    v16bf a = lds_load_A(sH, 64, row0, kc * 32);
#pragma unroll
    for (int nt = 0; nt < 4; ++nt) {
      v16bf b = lds_load_Bt(sRt, 64, kc * 32, nt * 16);
      acc[nt] = WMMA_BF16(a, b, acc[nt]);
    }
  }
#pragma unroll
  for (int nt = 0; nt < 4; ++nt)
#pragma unroll
    for (int i = 0; i < 8; ++i)
      sTt[(nt * 16 + col) * 64 + row0 + hb + i] = (bf16)acc[nt][i];  // transposed store
  __syncthreads();

  // out = Lt @ T
  v8f acc2[4] = {};
#pragma unroll
  for (int kc = 0; kc < 2; ++kc) {
    v16bf a = lds_load_A(sL, 64, row0, kc * 32);
#pragma unroll
    for (int nt = 0; nt < 4; ++nt) {
      v16bf b = lds_load_Bt(sTt, 64, kc * 32, nt * 16);
      acc2[nt] = WMMA_BF16(a, b, acc2[nt]);
    }
  }
  bf16* op = out + (size_t)tok * DM;
#pragma unroll
  for (int nt = 0; nt < 4; ++nt)
#pragma unroll
    for (int i = 0; i < 8; ++i)
      op[(row0 + hb + i) * 64 + nt * 16 + col] = (bf16)acc2[nt][i];
}

// ---------------- tiled bf16 GEMM: C(MxN f32) = A(MxK) @ Bt(NxK, N-major) ------
// 128x128 block, 256 threads (8 waves, each 32x64), double-buffered LDS with
// CDNA5 async global->LDS staging (ASYNCcnt).
__global__ __launch_bounds__(256) void k_gemm_bf16(const bf16* __restrict__ A,
                                                   const bf16* __restrict__ Bt,
                                                   float* __restrict__ C,
                                                   int M, int N, int K) {
  __shared__ __align__(16) bf16 sA[2][128 * 32];   // row-major [m][k]
  __shared__ __align__(16) bf16 sBt[2][128 * 32];  // N-major  [n][k]
  int bm = blockIdx.x * 128, bn = blockIdx.y * 128;
  int w = threadIdx.x >> 5;
  int wm = (w >> 1) * 32;       // 4 wave-rows
  int wn = (w & 1) * 64;        // 2 wave-cols
  v8f acc[2][4] = {};

  // async-stage K-chunk k0 into buffer p (2 b128 chunks each for A and Bt/thread)
  auto stage = [&](int k0, int p) {
#pragma unroll
    for (int i = threadIdx.x; i < 512; i += 256) {
      int r = i >> 2, c8 = (i & 3) * 8;
      async_load_b128(lds_off32(&sA[p][r * 32 + c8]),
                      &A[(size_t)(bm + r) * K + k0 + c8]);
      async_load_b128(lds_off32(&sBt[p][r * 32 + c8]),
                      &Bt[(size_t)(bn + r) * K + k0 + c8]);
    }
  };

  stage(0, 0);
  wait_asynccnt0();
  __syncthreads();
  int p = 0;
  for (int k0 = 0; k0 < K; k0 += 32) {
    if (k0 + 32 < K) {
      stage(k0 + 32, p ^ 1);
      if (k0 + 64 < K)
        __builtin_prefetch(&A[(size_t)(bm + (threadIdx.x >> 1)) * K + k0 + 64], 0, 3);
    }
    v16bf a0 = lds_load_A(sA[p], 32, wm, 0);
    v16bf a1 = lds_load_A(sA[p], 32, wm + 16, 0);
#pragma unroll
    for (int ni = 0; ni < 4; ++ni) {
      v16bf b = lds_load_Bt(sBt[p], 32, 0, wn + ni * 16);
      acc[0][ni] = WMMA_BF16(a0, b, acc[0][ni]);
      acc[1][ni] = WMMA_BF16(a1, b, acc[1][ni]);
    }
    wait_asynccnt0();   // next buffer's async stores have landed in LDS
    __syncthreads();
    p ^= 1;
  }
  int lane = threadIdx.x & 31;
  int col = lane & 15;
  int hb = (lane & 16) ? 8 : 0;
#pragma unroll
  for (int mi = 0; mi < 2; ++mi)
#pragma unroll
    for (int ni = 0; ni < 4; ++ni)
#pragma unroll
      for (int i = 0; i < 8; ++i)
        C[(size_t)(bm + wm + mi * 16 + hb + i) * N + bn + wn + ni * 16 + col] = acc[mi][ni][i];
}

// ---------------- RoPE in place on (S, NH*128) f32 ----------------
__global__ void k_rope(float* __restrict__ buf, const int* __restrict__ pos_ids, int NH) {
  int idx = blockIdx.x * blockDim.x + threadIdx.x;
  int total = S_LEN * NH * 64;
  if (idx >= total) return;
  int i = idx & 63;
  int tmp = idx >> 6;
  int h = tmp % NH;
  int t = tmp / NH;
  float pos = (float)pos_ids[t];
  float freq = pos * __powf(10000.0f, -(float)i / 64.0f);
  float sn, cs;
  __sincosf(freq, &sn, &cs);
  size_t base = (size_t)t * (NH * HD) + h * HD;
  float x1 = buf[base + i], x2 = buf[base + i + 64];
  buf[base + i]      = x1 * cs - x2 * sn;
  buf[base + i + 64] = x2 * cs + x1 * sn;
}

// ---------------- per-head 128x128 transform + optional fake-quant ------------
__global__ __launch_bounds__(128) void k_headT(const float* __restrict__ in,
                                               const float* __restrict__ T,
                                               bf16* __restrict__ out,
                                               int NHld, int transB, int bits, float scale) {
  __shared__ __align__(16) bf16 sA[64 * 128];    // row-major [m][k]
  __shared__ __align__(16) bf16 sBt[128 * 128];  // N-major  [n][k]
  int tb = blockIdx.x;
  int nh = blockIdx.y;
  const float* ip = in + (size_t)tb * 64 * NHld + nh * HD;
  for (int i = threadIdx.x; i < 64 * 128; i += 128) {
    int r = i >> 7, c = i & 127;
    sA[i] = (bf16)ip[(size_t)r * NHld + c];
  }
  for (int i = threadIdx.x; i < 128 * 128; i += 128) {
    int n = i >> 7, k = i & 127;
    sBt[i] = (bf16)(transB ? T[i] : T[k * 128 + n]);  // sBt[n][k] = B[k][n]
  }
  __syncthreads();
  int w = threadIdx.x >> 5;
  int row0 = w * 16;
  int lane = threadIdx.x & 31;
  int col = lane & 15;
  int hb = (lane & 16) ? 8 : 0;

  v8f acc[8] = {};
#pragma unroll
  for (int kc = 0; kc < 4; ++kc) {
    v16bf a = lds_load_A(sA, 128, row0, kc * 32);
#pragma unroll
    for (int nt = 0; nt < 8; ++nt) {
      v16bf b = lds_load_Bt(sBt, 128, kc * 32, nt * 16);
      acc[nt] = WMMA_BF16(a, b, acc[nt]);
    }
  }

  float s[8];
  float qmax = 1.0f;
  if (bits > 0) {
    qmax = (float)((1 << (bits - 1)) - 1);
    float amax[8];
#pragma unroll
    for (int i = 0; i < 8; ++i) {
      amax[i] = 0.0f;
#pragma unroll
      for (int nt = 0; nt < 8; ++nt) amax[i] = fmaxf(amax[i], fabsf(acc[nt][i] * scale));
    }
#pragma unroll
    for (int off = 1; off < 16; off <<= 1)
#pragma unroll
      for (int i = 0; i < 8; ++i) amax[i] = fmaxf(amax[i], __shfl_xor(amax[i], off, 32));
#pragma unroll
    for (int i = 0; i < 8; ++i) s[i] = fmaxf(amax[i] / qmax, 1e-8f);
  }
#pragma unroll
  for (int nt = 0; nt < 8; ++nt)
#pragma unroll
    for (int i = 0; i < 8; ++i) {
      float v = acc[nt][i] * scale;
      if (bits > 0) {
        float q = rintf(v / s[i]);
        q = fminf(fmaxf(q, -qmax), qmax);
        v = q * s[i];
      }
      out[(size_t)(tb * 64 + row0 + hb + i) * NHld + nh * HD + nt * 16 + col] = (bf16)v;
    }
}

// ---------------- flash attention (causal, GQA), q pre-scaled by 1/sqrt(HD) ---
__global__ __launch_bounds__(128) void k_attn(const bf16* __restrict__ q2,
                                              const bf16* __restrict__ k2,
                                              const bf16* __restrict__ v2,
                                              float* __restrict__ o) {
  __shared__ __align__(16) bf16 sK[64 * 128];   // [kv][d]  (N-major for Q.K^T)
  __shared__ __align__(16) bf16 sVt[128 * 64];  // [d][kv]  (N-major for P.V)
  __shared__ __align__(16) bf16 sP[64 * 64];    // row-major [q][kv] (A side)
  int qb = blockIdx.x;
  int head = blockIdx.y;
  int kvh = head >> 2;  // H/KVH = 4
  int w = threadIdx.x >> 5;
  int row0 = w * 16;
  int lane = threadIdx.x & 31;
  int col = lane & 15;
  int hb = (lane & 16) ? 8 : 0;

  // stage Q tile through sK, keep fragments in registers
  for (int i = threadIdx.x; i < 1024; i += 128) {
    int r = i >> 4, d8 = (i & 15) * 8;
    *(v8bf*)&sK[r * 128 + d8] =
        *(const v8bf*)&q2[(size_t)(qb * 64 + r) * (NHEAD * HD) + head * HD + d8];
  }
  __syncthreads();
  v16bf qf[4];
#pragma unroll
  for (int kc = 0; kc < 4; ++kc) qf[kc] = lds_load_A(sK, 128, row0, kc * 32);

  v8f acc[8] = {};
  float m[8], l[8];
#pragma unroll
  for (int i = 0; i < 8; ++i) { m[i] = -1e30f; l[i] = 0.0f; }

  for (int kb = 0; kb <= qb; ++kb) {
    __syncthreads();
    for (int i = threadIdx.x; i < 1024; i += 128) {
      int r = i >> 4, d8 = (i & 15) * 8;
      size_t gi = (size_t)(kb * 64 + r) * (NKVH * HD) + kvh * HD + d8;
      *(v8bf*)&sK[r * 128 + d8] = *(const v8bf*)&k2[gi];
      v8bf vv = *(const v8bf*)&v2[gi];
#pragma unroll
      for (int j = 0; j < 8; ++j) sVt[(d8 + j) * 64 + r] = vv[j];  // transpose V
    }
    __syncthreads();

    // scores S = Q @ K^T  (64x64 per block; 16x64 per wave)
    v8f sc[4];
#pragma unroll
    for (int nt = 0; nt < 4; ++nt) {
      v8f c = {};
#pragma unroll
      for (int kc = 0; kc < 4; ++kc) {
        v16bf b = lds_load_Bt(sK, 128, kc * 32, nt * 16);
        c = WMMA_BF16(qf[kc], b, c);
      }
      sc[nt] = c;
    }
    if (kb == qb) {  // diagonal block: causal mask (matches reference's -1e9)
#pragma unroll
      for (int nt = 0; nt < 4; ++nt)
#pragma unroll
        for (int i = 0; i < 8; ++i) {
          int rg = row0 + hb + i;
          int cg = nt * 16 + col;
          if (cg > rg) sc[nt][i] += -1e9f;
        }
    }

    // online softmax: row stats via 16-lane butterfly within each half-wave
    float mn[8];
#pragma unroll
    for (int i = 0; i < 8; ++i) {
      mn[i] = m[i];
#pragma unroll
      for (int nt = 0; nt < 4; ++nt) mn[i] = fmaxf(mn[i], sc[nt][i]);
    }
#pragma unroll
    for (int off = 1; off < 16; off <<= 1)
#pragma unroll
      for (int i = 0; i < 8; ++i) mn[i] = fmaxf(mn[i], __shfl_xor(mn[i], off, 32));

    float corr[8], rs[8];
#pragma unroll
    for (int i = 0; i < 8; ++i) {
      corr[i] = __expf(m[i] - mn[i]);
      m[i] = mn[i];
      rs[i] = 0.0f;
    }
#pragma unroll
    for (int nt = 0; nt < 4; ++nt)
#pragma unroll
      for (int i = 0; i < 8; ++i) {
        float p = __expf(sc[nt][i] - m[i]);
        rs[i] += p;
        sP[(row0 + hb + i) * 64 + nt * 16 + col] = (bf16)p;  // wave-private rows
      }
#pragma unroll
    for (int off = 1; off < 16; off <<= 1)
#pragma unroll
      for (int i = 0; i < 8; ++i) rs[i] += __shfl_xor(rs[i], off, 32);
#pragma unroll
    for (int i = 0; i < 8; ++i) l[i] = l[i] * corr[i] + rs[i];
#pragma unroll
    for (int vt = 0; vt < 8; ++vt)
#pragma unroll
      for (int i = 0; i < 8; ++i) acc[vt][i] *= corr[i];

    // acc += P @ V   (LDS DS ops are in-order per wave; sP rows are wave-private)
    v16bf pf0 = lds_load_A(sP, 64, row0, 0);
    v16bf pf1 = lds_load_A(sP, 64, row0, 32);
#pragma unroll
    for (int vt = 0; vt < 8; ++vt) {
      v16bf b0 = lds_load_Bt(sVt, 64, 0, vt * 16);
      acc[vt] = WMMA_BF16(pf0, b0, acc[vt]);
      v16bf b1 = lds_load_Bt(sVt, 64, 32, vt * 16);
      acc[vt] = WMMA_BF16(pf1, b1, acc[vt]);
    }
  }

#pragma unroll
  for (int vt = 0; vt < 8; ++vt)
#pragma unroll
    for (int i = 0; i < 8; ++i)
      o[(size_t)(qb * 64 + row0 + hb + i) * (NHEAD * HD) + head * HD + vt * 16 + col] =
          acc[vt][i] / l[i];
}

// ---------------- host orchestration ----------------
static inline size_t align256(size_t x) { return (x + 255) & ~(size_t)255; }

extern "C" void kernel_launch(void* const* d_in, const int* in_sizes, int n_in,
                              void* d_out, int out_size, void* d_ws, size_t ws_size,
                              hipStream_t stream) {
  (void)in_sizes; (void)n_in; (void)out_size; (void)ws_size;
  const float* hidden = (const float*)d_in[0];
  // d_in[1] = attention_mask (causal; applied analytically)
  const int*   pos_ids = (const int*)d_in[2];
  const float* ln_left = (const float*)d_in[3];
  const float* ln_right= (const float*)d_in[4];
  const float* Wq = (const float*)d_in[5];
  const float* Wk = (const float*)d_in[6];
  const float* Wv = (const float*)d_in[7];
  const float* Wo = (const float*)d_in[8];
  const float* Tk = (const float*)d_in[9];
  const float* Tv = (const float*)d_in[10];
  float* out = (float*)d_out;

  // workspace carve-up
  char* p = (char*)d_ws;
  size_t off = 0;
  auto carve = [&](size_t bytes) { void* r = p + off; off += align256(bytes); return r; };
  bf16* Wq_t = (bf16*)carve((size_t)DM * DM * 2);             // N-major [n][k]
  bf16* Wk_t = (bf16*)carve((size_t)DM * NKVH * HD * 2);
  bf16* Wv_t = (bf16*)carve((size_t)DM * NKVH * HD * 2);
  bf16* Wo_t = (bf16*)carve((size_t)DM * DM * 2);
  bf16* Rt_bf = (bf16*)carve(64 * 64 * 2);
  bf16* Lt_bf = (bf16*)carve(64 * 64 * 2);
  float* Tk_inv = (float*)carve(128 * 128 * 4);
  float* Tv_inv = (float*)carve(128 * 128 * 4);
  float* gj_work = (float*)carve(128 * 128 * 4);
  bf16* hprime = (bf16*)carve((size_t)S_LEN * DM * 2);
  float* q_raw = (float*)carve((size_t)S_LEN * DM * 4);      // later reused as o_raw
  float* k_raw = (float*)carve((size_t)S_LEN * NKVH * HD * 4);
  float* v_raw = (float*)carve((size_t)S_LEN * NKVH * HD * 4);
  bf16* q2 = (bf16*)carve((size_t)S_LEN * DM * 2);
  bf16* k2 = (bf16*)carve((size_t)S_LEN * NKVH * HD * 2);
  bf16* v2 = (bf16*)carve((size_t)S_LEN * NKVH * HD * 2);
  float* o_raw = q_raw;     // alias: q_raw dead after q2 is built
  bf16* o2 = hprime;        // alias: hprime dead after projections

  // 1) weight conversions to bf16, pre-transposed to N-major for the GEMMs
  {
    size_t n;
    n = (size_t)DM * DM;
    k_f2bfT<<<(unsigned)((n + 255) / 256), 256, 0, stream>>>(Wq, Wq_t, DM, DM);
    k_f2bfT<<<(unsigned)((n + 255) / 256), 256, 0, stream>>>(Wo, Wo_t, DM, DM);
    n = (size_t)DM * NKVH * HD;
    k_f2bfT<<<(unsigned)((n + 255) / 256), 256, 0, stream>>>(Wk, Wk_t, DM, NKVH * HD);
    k_f2bfT<<<(unsigned)((n + 255) / 256), 256, 0, stream>>>(Wv, Wv_t, DM, NKVH * HD);
    k_f2bfT<<<16, 256, 0, stream>>>(ln_right, Rt_bf, 64, 64);  // R^T, N-major B tile
    k_f2bfT<<<16, 256, 0, stream>>>(ln_left, Lt_bf, 64, 64);   // L^T, A side
  }

  // 2) invert Tk, Tv (serialized on stream; reuse scratch)
  k_invert128<<<1, 128, 0, stream>>>(Tk, Tk_inv, gj_work);
  k_invert128<<<1, 128, 0, stream>>>(Tv, Tv_inv, gj_work);

  // 3) per-token 64x64 transform -> hprime (bf16)
  k_token_transform<<<S_LEN, 128, 0, stream>>>(hidden, Lt_bf, Rt_bf, hprime);

  // 4) QKV projections (f32 outputs)
  k_gemm_bf16<<<dim3(S_LEN / 128, DM / 128), 256, 0, stream>>>(hprime, Wq_t, q_raw, S_LEN, DM, DM);
  k_gemm_bf16<<<dim3(S_LEN / 128, (NKVH * HD) / 128), 256, 0, stream>>>(hprime, Wk_t, k_raw, S_LEN, NKVH * HD, DM);
  k_gemm_bf16<<<dim3(S_LEN / 128, (NKVH * HD) / 128), 256, 0, stream>>>(hprime, Wv_t, v_raw, S_LEN, NKVH * HD, DM);

  // 5) RoPE on q, k
  k_rope<<<(S_LEN * NHEAD * 64 + 255) / 256, 256, 0, stream>>>(q_raw, pos_ids, NHEAD);
  k_rope<<<(S_LEN * NKVH * 64 + 255) / 256, 256, 0, stream>>>(k_raw, pos_ids, NKVH);

  // 6) per-head transforms (+ fake-quant for k, v; 1/sqrt(HD) folded into q)
  // q @ inv(Tk)^T: B[k][n] = inv[n][k] -> sBt[n][k] = inv[n*128+k] (transB=1 path)
  const float inv_sqrt_hd = 0.08838834764831845f;  // 1/sqrt(128)
  k_headT<<<dim3(S_LEN / 64, NHEAD), 128, 0, stream>>>(q_raw, Tk_inv, q2, NHEAD * HD, /*transB=*/1, /*bits=*/0, inv_sqrt_hd);
  k_headT<<<dim3(S_LEN / 64, NKVH), 128, 0, stream>>>(k_raw, Tk, k2, NKVH * HD, 0, /*bits=*/4, 1.0f);
  k_headT<<<dim3(S_LEN / 64, NKVH), 128, 0, stream>>>(v_raw, Tv, v2, NKVH * HD, 0, /*bits=*/4, 1.0f);

  // 7) causal GQA flash attention -> o_raw (f32, (S, H*HD))
  k_attn<<<dim3(S_LEN / 64, NHEAD), 128, 0, stream>>>(q2, k2, v2, o_raw);

  // 8) o @ Tv_inv per head -> o2 (bf16)
  k_headT<<<dim3(S_LEN / 64, NHEAD), 128, 0, stream>>>(o_raw, Tv_inv, o2, NHEAD * HD, 0, /*bits=*/0, 1.0f);

  // 9) final projection: out = o2 @ Wo (f32)
  k_gemm_bf16<<<dim3(S_LEN / 128, DM / 128), 256, 0, stream>>>(o2, Wo_t, out, S_LEN, DM, DM);
}
